// SPADE_LeWinTransformerBlock_29480655520178
// MI455X (gfx1250) — matrix operand order repaired and checked by reading, not compiled, not run on hardware
//
#include <hip/hip_runtime.h>
#include <hip/hip_bf16.h>
#include <stdint.h>

// ---------------- problem constants ----------------
#define B_     8
#define HS     128
#define WS_    128
#define C_     192
#define NHEAD  6
#define WIN_   8
#define SHIFT_ 4
#define L_     (HS*WS_)     // 16384
#define HID_   768
#define DH     32           // C/NH
#define NW     64           // tokens per window

// ---------------- WMMA plumbing ----------------
typedef __attribute__((ext_vector_type(16))) __bf16 v16bf;
typedef __attribute__((ext_vector_type(8)))  float  v8f;
typedef __attribute__((ext_vector_type(4)))  int    i32x4;

union ABf { uint32_t u[8]; v16bf v; };

__device__ __forceinline__ float bf2f(uint16_t h){
  union { uint32_t u; float f; } c; c.u = ((uint32_t)h) << 16; return c.f;
}
__device__ __forceinline__ uint16_t f2bf(float f){
  union { float f; uint32_t u; } c; c.f = f;
  uint32_t u = c.u + 0x7FFFu + ((c.u >> 16) & 1u);   // round-to-nearest-even
  return (uint16_t)(u >> 16);
}
__device__ __forceinline__ float gelu_f(float x){
  return 0.5f * x * (1.f + erff(x * 0.70710678118654752f));
}
__device__ __forceinline__ v8f wmma_bf(ABf a, ABf b, v8f c){
  return __builtin_amdgcn_wmma_f32_16x16x32_bf16(false, a.v, false, b.v, (short)0, c, false, false);
}
__device__ __forceinline__ int reg3(int v){ return v < HS - WIN_ ? 0 : (v < HS - SHIFT_ ? 1 : 2); }

// ---------------- CDNA5 async global->LDS (ASYNCcnt) ----------------
// Guarded: falls back to plain LDS stores if the builtins are absent.
#if defined(__gfx1250__) && \
    __has_builtin(__builtin_amdgcn_global_load_async_to_lds_b128) && \
    __has_builtin(__builtin_amdgcn_s_wait_asynccnt)
#define ASYNC_LDS 1
#else
#define ASYNC_LDS 0
#endif

__device__ __forceinline__ void cp16_g2l(void* lds, const void* g) {
#if ASYNC_LDS
  __builtin_amdgcn_global_load_async_to_lds_b128(
      (__attribute__((address_space(1))) i32x4*)(g),
      (__attribute__((address_space(3))) i32x4*)(lds), 0, 0);
#else
  *(uint4*)lds = *(const uint4*)g;
#endif
}
__device__ __forceinline__ void async_wait0() {
#if ASYNC_LDS
  __builtin_amdgcn_s_wait_asynccnt(0);
#endif
}
__device__ __forceinline__ void async_wait2() {
#if ASYNC_LDS
  __builtin_amdgcn_s_wait_asynccnt(2);
#endif
}

// =====================================================================
// Weight packing: f32 -> bf16, k-pair-interleaved so B-frag loads are b32
//   dst_pair_u32[(k>>1)*N + n] = { B[k][n], B[k+1][n] }
// =====================================================================
__global__ void pack_rowmajor(const float* __restrict__ src, uint16_t* __restrict__ dst,
                              int K, int N) {
  int i = blockIdx.x * 256 + threadIdx.x;
  if (i >= K * N) return;
  int k = i / N, n = i % N;
  dst[((size_t)(k >> 1) * N + n) * 2 + (k & 1)] = f2bf(src[(size_t)k * N + n]);
}

// conv weight OIHW (192,128,3,3) -> GEMM B with K = tap*128 + cin, N = cout
__global__ void pack_conv(const float* __restrict__ w, uint16_t* __restrict__ dst) {
  int i = blockIdx.x * 256 + threadIdx.x;
  if (i >= 1152 * 192) return;
  int k = i / 192, n = i % 192;
  int tap = k >> 7, cin = k & 127;
  float v = w[(size_t)(n * 128 + cin) * 9 + tap];
  dst[((size_t)(k >> 1) * 192 + n) * 2 + (k & 1)] = f2bf(v);
}

// =====================================================================
// SPADE stage-1 conv: 1 -> 128 channels, 3x3 SAME, leaky_relu(0.01)
// src (B,1,128,128) f32 -> a (B, H*W, 128) NHWC bf16
// =====================================================================
__global__ __launch_bounds__(256) void conv1_lrelu(const float* __restrict__ src,
                                                   const float* __restrict__ w,
                                                   const float* __restrict__ bias,
                                                   uint16_t* __restrict__ a) {
  __shared__ float sw[128 * 9];
  __shared__ float sb[128];
  for (int i = threadIdx.x; i < 128 * 9; i += 256) sw[i] = w[i];
  for (int i = threadIdx.x; i < 128; i += 256) sb[i] = bias[i];
  __syncthreads();
  int pix  = blockIdx.x * 128 + (threadIdx.x >> 1);
  int part = threadIdx.x & 1;
  int b = pix >> 14, y = (pix >> 7) & 127, x = pix & 127;
  float v9[9];
  #pragma unroll
  for (int t = 0; t < 9; ++t) {
    int yy = y + t / 3 - 1, xx = x + t % 3 - 1;
    v9[t] = (yy >= 0 && yy < 128 && xx >= 0 && xx < 128)
              ? src[(size_t)(b << 14) + (yy << 7) + xx] : 0.f;
  }
  for (int co = part * 64; co < part * 64 + 64; ++co) {
    float s = sb[co];
    #pragma unroll
    for (int t = 0; t < 9; ++t) s += sw[co * 9 + t] * v9[t];
    s = s > 0.f ? s : 0.01f * s;
    a[(size_t)pix * 128 + co] = f2bf(s);
  }
}

// =====================================================================
// gamma/beta conv 128 -> 192, 3x3 SAME as implicit GEMM (K = 1152)
// 4 waves; wave owns 3 N-tiles x 4 M-tiles (B frags loaded once per wave)
// halo tile staged via async global->LDS
// =====================================================================
__global__ __launch_bounds__(128) void conv_gb(const uint16_t* __restrict__ a,
                                               const uint16_t* __restrict__ wp,
                                               const float* __restrict__ bias,
                                               uint16_t* __restrict__ outb) {
  __shared__ __align__(16) uint16_t halo[10 * 10 * 128];   // 25600 B
  int blk = blockIdx.x;
  int b  = blk >> 8;
  int by = (blk >> 4) & 15;
  int bx = blk & 15;
  const uint32_t* a32 = (const uint32_t*)a;
  // stage halo: 1600 chunks of 16B (one pixel = 16 chunks)
  for (int i = threadIdx.x; i < 1600; i += 128) {
    int hp = i >> 4, c16 = i & 15;
    int hy = by * 8 + hp / 10 - 1;
    int hx = bx * 8 + hp % 10 - 1;
    void* dst = (void*)(halo + hp * 128 + c16 * 8);
    if (hy >= 0 && hy < 128 && hx >= 0 && hx < 128) {
      cp16_g2l(dst, (const void*)(a32 + (size_t)((b << 14) + (hy << 7) + hx) * 64 + c16 * 4));
    } else {
      uint4 z = {0u, 0u, 0u, 0u};
      *(uint4*)dst = z;
    }
  }
  async_wait0();
  __syncthreads();

  int wave = threadIdx.x >> 5, lane = threadIdx.x & 31;
  int l15 = lane & 15, hi = lane >> 4;
  int kh2 = hi * 8, kbh2 = hi * 8;
  v8f zero = {0.f,0.f,0.f,0.f,0.f,0.f,0.f,0.f};
  v8f acc[4][3];
  #pragma unroll
  for (int mt = 0; mt < 4; ++mt)
    #pragma unroll
    for (int nt = 0; nt < 3; ++nt) acc[mt][nt] = zero;

  const uint32_t* h32  = (const uint32_t*)halo;
  const uint32_t* wp32 = (const uint32_t*)wp;
  for (int ks = 0; ks < 36; ++ks) {          // K = 1152 = 36 * 32
    int k0  = ks << 5;
    int tap = k0 >> 7;
    int c2  = (k0 & 127) >> 1;
    int dy = tap / 3, dx = tap % 3;
    ABf af[4];
    #pragma unroll
    for (int mt = 0; mt < 4; ++mt) {
      int m = mt * 16 + l15;
      const uint32_t* ap = h32 + (size_t)(((m >> 3) + dy) * 10 + ((m & 7) + dx)) * 64 + c2;
      #pragma unroll
      for (int j = 0; j < 4; ++j) { af[mt].u[j] = ap[kh2 + j]; af[mt].u[4 + j] = ap[kh2 + 4 + j]; }
    }
    const uint32_t* bp = wp32 + (size_t)(k0 >> 1) * 192;
    #pragma unroll
    for (int nt = 0; nt < 3; ++nt) {
      ABf bf;
      int n = (wave * 3 + nt) * 16 + l15;
      #pragma unroll
      for (int v = 0; v < 8; ++v) bf.u[v] = bp[(size_t)(kbh2 + v) * 192 + n];
      #pragma unroll
      for (int mt = 0; mt < 4; ++mt) acc[mt][nt] = wmma_bf(af[mt], bf, acc[mt][nt]);
    }
  }
  #pragma unroll
  for (int nt = 0; nt < 3; ++nt) {
    int n = (wave * 3 + nt) * 16 + l15;
    float bv = bias[n];
    #pragma unroll
    for (int mt = 0; mt < 4; ++mt) {
      int mb = mt * 16 + hi * 8;
      #pragma unroll
      for (int r = 0; r < 8; ++r) {
        int mm = mb + r;
        int py = by * 8 + (mm >> 3), px = bx * 8 + (mm & 7);
        outb[(size_t)((b << 14) + (py << 7) + px) * 192 + n] = f2bf(acc[mt][nt][r] + bv);
      }
    }
  }
}

// =====================================================================
// Fused LayerNorm + SPADE modulation; optional shift+window-partition
// =====================================================================
__global__ __launch_bounds__(256) void spade_mod(const float* __restrict__ xin,
                                                 const uint16_t* __restrict__ gamma,
                                                 const uint16_t* __restrict__ beta,
                                                 const float* __restrict__ lng,
                                                 const float* __restrict__ lnb,
                                                 uint16_t* __restrict__ outb,
                                                 int windowed) {
  int tok  = blockIdx.x * 8 + (threadIdx.x >> 5);
  int lane = threadIdx.x & 31;
  const float* xp = xin + (size_t)tok * 192;
  float v[6], s = 0.f;
  #pragma unroll
  for (int j = 0; j < 6; ++j) { v[j] = xp[lane + j * 32]; s += v[j]; }
  #pragma unroll
  for (int o = 16; o; o >>= 1) s += __shfl_xor(s, o, 32);
  float mu = s * (1.f / 192.f);
  float var = 0.f;
  #pragma unroll
  for (int j = 0; j < 6; ++j) { float d = v[j] - mu; var += d * d; }
  #pragma unroll
  for (int o = 16; o; o >>= 1) var += __shfl_xor(var, o, 32);
  float rstd = rsqrtf(var * (1.f / 192.f) + 1e-5f);
  int b = tok >> 14, l = tok & 16383;
  size_t obase;
  if (windowed) {
    int y = l >> 7, x = l & 127;
    int ys = (y - SHIFT_) & 127, xs = (x - SHIFT_) & 127;     // roll(-SHIFT)
    int w = (ys >> 3) * 16 + (xs >> 3);
    int p = ((ys & 7) << 3) + (xs & 7);
    obase = (((size_t)b * 256 + w) * 64 + p) * 192;
  } else {
    obase = (size_t)tok * 192;
  }
  #pragma unroll
  for (int j = 0; j < 6; ++j) {
    int c = lane + j * 32;
    float n  = (v[j] - mu) * rstd * lng[c] + lnb[c];
    float gm = bf2f(gamma[(size_t)tok * 192 + c]);
    float bt = bf2f(beta [(size_t)tok * 192 + c]);
    outb[obase + c] = f2bf(n * (1.f + gm) + bt);
  }
}

// =====================================================================
// Windowed attention: one workgroup (8 waves) per window (2048 windows)
// =====================================================================
__global__ __launch_bounds__(256) void attn_win(const uint16_t* __restrict__ xw,
                                                const uint16_t* __restrict__ qkvp,
                                                const float* __restrict__ qkv_b,
                                                const float* __restrict__ rpb,
                                                const uint16_t* __restrict__ projp,
                                                const float* __restrict__ proj_b,
                                                const float* __restrict__ xres,
                                                float* __restrict__ outp) {
  __shared__ __align__(16) uint16_t sx[NW * 192];   // xw tile, reused as O (bf16)
  __shared__ __align__(16) uint16_t sq[NW * 192];
  __shared__ __align__(16) uint16_t sk[NW * 192];
  __shared__ __align__(16) uint16_t sv[NW * 192];
  __shared__ __align__(16) float    ss[NW * NW];
  __shared__ __align__(16) uint16_t sp[NW * NW];

  int widx = blockIdx.x;                  // b*256 + wy*16 + wx
  int b = widx >> 8, w = widx & 255;
  int wy = w >> 4, wx = w & 15;
  int wave = threadIdx.x >> 5, lane = threadIdx.x & 31;
  int l15 = lane & 15, hi = lane >> 4;
  int kh2 = hi * 8, kbh2 = hi * 8;
  v8f zero = {0.f,0.f,0.f,0.f,0.f,0.f,0.f,0.f};

  // stage input tile (24576 B = 1536 x 16B) via async copy
  const char* xsrc = (const char*)(xw + (size_t)widx * NW * 192);
  for (int i = threadIdx.x; i < 1536; i += 256)
    cp16_g2l((void*)(sx + i * 8), (const void*)(xsrc + (size_t)i * 16));
  async_wait0();
  __syncthreads();

  uint32_t* sx32 = (uint32_t*)sx;

  // ---- qkv: (64x192)@(192x576): wave owns 2 M-tiles x 9 N-tiles ----
  const uint32_t* qkv32 = (const uint32_t*)qkvp;
  {
    int mt0 = (wave & 1) * 2;
    int nt0 = (wave >> 1) * 9;
    v8f accq[2][9];
    #pragma unroll
    for (int p = 0; p < 2; ++p)
      #pragma unroll
      for (int j = 0; j < 9; ++j) accq[p][j] = zero;
    for (int ks = 0; ks < 6; ++ks) {
      ABf af[2];
      #pragma unroll
      for (int p = 0; p < 2; ++p) {
        const uint32_t* ap = sx32 + (size_t)((mt0 + p) * 16 + l15) * 96 + ks * 16;
        #pragma unroll
        for (int j = 0; j < 4; ++j) { af[p].u[j] = ap[kh2 + j]; af[p].u[4 + j] = ap[kh2 + 4 + j]; }
      }
      const uint32_t* bp = qkv32 + (size_t)(ks * 16) * 576;
      #pragma unroll
      for (int j = 0; j < 9; ++j) {
        ABf bf;
        int n = (nt0 + j) * 16 + l15;
        #pragma unroll
        for (int v = 0; v < 8; ++v) bf.u[v] = bp[(size_t)(kbh2 + v) * 576 + n];
        #pragma unroll
        for (int p = 0; p < 2; ++p) accq[p][j] = wmma_bf(af[p], bf, accq[p][j]);
      }
    }
    #pragma unroll
    for (int j = 0; j < 9; ++j) {
      int n = (nt0 + j) * 16 + l15;
      float bv = qkv_b[n];
      #pragma unroll
      for (int p = 0; p < 2; ++p) {
        int mb = (mt0 + p) * 16 + hi * 8;
        #pragma unroll
        for (int r = 0; r < 8; ++r) {
          int m = mb + r;
          float val = accq[p][j][r] + bv;
          if (n < 192)      sq[(size_t)m * 192 + n] = f2bf(val * 0.17677669529663687f); // d^-0.5
          else if (n < 384) { int c = n - 192; sk[((size_t)(c >> 1) * 64 + m) * 2 + (c & 1)] = f2bf(val); }
          else              { int c = n - 384; sv[((size_t)(m >> 1) * 192 + c) * 2 + (m & 1)] = f2bf(val); }
        }
      }
    }
  }
  __syncthreads();

  const uint32_t* sq32 = (const uint32_t*)sq;
  const uint32_t* sk32 = (const uint32_t*)sk;
  const uint32_t* sv32 = (const uint32_t*)sv;
  const uint32_t* sp32 = (const uint32_t*)sp;

  for (int h = 0; h < NHEAD; ++h) {
    // ---- S = Qh @ Kh^T : 16 tiles, 2 per wave ----
    for (int i = 0; i < 2; ++i) {
      int t = wave * 2 + i;
      int mt = t >> 2, nt = t & 3;
      v8f acc = zero;
      ABf af, bf;
      int row = mt * 16 + l15;
      int n   = nt * 16 + l15;
      const uint32_t* ap = sq32 + (size_t)row * 96 + h * 16;
      #pragma unroll
      for (int j = 0; j < 4; ++j) { af.u[j] = ap[kh2 + j]; af.u[4 + j] = ap[kh2 + 4 + j]; }
      const uint32_t* bp = sk32 + (size_t)(h * 16) * 64;
      #pragma unroll
      for (int v = 0; v < 8; ++v) bf.u[v] = bp[(size_t)(kbh2 + v) * 64 + n];
      acc = wmma_bf(af, bf, acc);
      int mb = mt * 16 + hi * 8;
      int ny = n >> 3, nx = n & 7;
      int lnk = reg3(wy * 8 + ny) * 3 + reg3(wx * 8 + nx);
      #pragma unroll
      for (int r = 0; r < 8; ++r) {
        int m = mb + r;
        int my = m >> 3, mx = m & 7;
        int rel = (my - ny + 7) * 15 + (mx - nx + 7);
        int lm  = reg3(wy * 8 + my) * 3 + reg3(wx * 8 + mx);
        float mk = (lm != lnk) ? -100.f : 0.f;
        ss[(size_t)m * 64 + n] = acc[r] + rpb[rel * NHEAD + h] + mk;
      }
    }
    __syncthreads();
    // ---- softmax rows: 4 adjacent lanes per row ----
    {
      int rrow = threadIdx.x >> 2, q = threadIdx.x & 3;
      float* rp = ss + (size_t)rrow * 64 + q * 16;
      float mx = -1e30f;
      #pragma unroll
      for (int j = 0; j < 16; ++j) mx = fmaxf(mx, rp[j]);
      mx = fmaxf(mx, __shfl_xor(mx, 1, 32));
      mx = fmaxf(mx, __shfl_xor(mx, 2, 32));
      float e[16], sm = 0.f;
      #pragma unroll
      for (int j = 0; j < 16; ++j) { e[j] = __expf(rp[j] - mx); sm += e[j]; }
      sm += __shfl_xor(sm, 1, 32);
      sm += __shfl_xor(sm, 2, 32);
      float inv = 1.f / sm;
      #pragma unroll
      for (int j = 0; j < 16; ++j) sp[(size_t)rrow * 64 + q * 16 + j] = f2bf(e[j] * inv);
    }
    __syncthreads();
    // ---- O_h = P @ V_h : 8 tiles, 1 per wave ----
    {
      int mt = wave >> 1, nt = wave & 1;
      v8f acc = zero;
      int row = mt * 16 + l15;
      int n   = nt * 16 + l15;
      for (int ks = 0; ks < 2; ++ks) {
        ABf af, bf;
        const uint32_t* ap = sp32 + (size_t)row * 32 + ks * 16;
        #pragma unroll
        for (int j = 0; j < 4; ++j) { af.u[j] = ap[kh2 + j]; af.u[4 + j] = ap[kh2 + 4 + j]; }
        #pragma unroll
        for (int v = 0; v < 8; ++v)
          bf.u[v] = sv32[(size_t)(ks * 16 + kbh2 + v) * 192 + h * 32 + n];
        acc = wmma_bf(af, bf, acc);
      }
      int mb = mt * 16 + hi * 8;
      #pragma unroll
      for (int r = 0; r < 8; ++r)
        sx[(size_t)(mb + r) * 192 + h * 32 + n] = f2bf(acc[r]);   // O into sx (bf16)
    }
    __syncthreads();
  }

  // ---- Y = O @ proj_w + proj_b ; window-reverse + roll(+SHIFT) + residual ----
  const uint32_t* pr32 = (const uint32_t*)projp;
  for (int i = 0; i < 6; ++i) {
    int t = wave + (i << 3);
    int mt = t & 3, nt = t >> 2;
    v8f acc = zero;
    int row = mt * 16 + l15;
    int n   = nt * 16 + l15;
    for (int ks = 0; ks < 6; ++ks) {
      ABf af, bf;
      const uint32_t* ap = sx32 + (size_t)row * 96 + ks * 16;
      #pragma unroll
      for (int j = 0; j < 4; ++j) { af.u[j] = ap[kh2 + j]; af.u[4 + j] = ap[kh2 + 4 + j]; }
      const uint32_t* bp = pr32 + (size_t)(ks * 16) * 192;
      #pragma unroll
      for (int v = 0; v < 8; ++v) bf.u[v] = bp[(size_t)(kbh2 + v) * 192 + n];
      acc = wmma_bf(af, bf, acc);
    }
    float bv = proj_b[n];
    int mb = mt * 16 + hi * 8;
    #pragma unroll
    for (int r = 0; r < 8; ++r) {
      int m = mb + r;
      int Y = wy * 8 + (m >> 3), X = wx * 8 + (m & 7);
      int y = (Y + SHIFT_) & 127, x = (X + SHIFT_) & 127;
      size_t idx = ((size_t)(b << 14) + (y << 7) + x) * 192 + n;
      outp[idx] = xres[idx] + acc[r] + bv;
    }
  }
}

// =====================================================================
// Generic bf16 WMMA GEMM with async-LDS double-buffered panel staging
// A[M][K] bf16 row-major, B packed pairs, bias
// mode 0: out_bf16 = gelu(A@B + bias)      (LeFF l1)
// mode 1: out_f32 += A@B + bias            (LeFF l2 residual into d_out)
// block = 256 threads, 64x64 tile; panels: A 64x32, B 32x64 (4KB each x2)
// =====================================================================
__global__ __launch_bounds__(256) void gemm_bf16(const uint16_t* __restrict__ A,
                                                 const uint16_t* __restrict__ Bp,
                                                 const float* __restrict__ bias,
                                                 uint16_t* __restrict__ obf,
                                                 float* __restrict__ of32,
                                                 int M, int N, int K, int mode) {
  __shared__ __align__(16) uint32_t sA[2][64 * 16];
  __shared__ __align__(16) uint32_t sB[2][16 * 64];
  int nBlkN = N >> 6;
  int bm = blockIdx.x / nBlkN, bn = blockIdx.x % nBlkN;
  int tid = threadIdx.x;
  int wave = tid >> 5, lane = tid & 31;
  int l15 = lane & 15, hi = lane >> 4;
  int kh2 = hi * 8, kbh2 = hi * 8;
  int K2 = K >> 1, nks = K >> 5;
  const uint32_t* A32 = (const uint32_t*)A;
  const uint32_t* B32 = (const uint32_t*)Bp;
  int arow = tid >> 2, ap4 = (tid & 3) * 4;   // A panel: 64 rows x 16 uints
  int bkp  = tid >> 4, bn4 = (tid & 15) * 4;  // B panel: 16 kp  x 64 uints

  auto stage = [&](int ks, int buf) {
    cp16_g2l((void*)&sA[buf][arow * 16 + ap4],
             (const void*)(A32 + (size_t)(bm * 64 + arow) * K2 + ks * 16 + ap4));
    cp16_g2l((void*)&sB[buf][bkp * 64 + bn4],
             (const void*)(B32 + (size_t)(ks * 16 + bkp) * N + bn * 64 + bn4));
  };

  stage(0, 0);
  v8f zero = {0.f,0.f,0.f,0.f,0.f,0.f,0.f,0.f};
  v8f acc[2]; acc[0] = zero; acc[1] = zero;

  for (int ks = 0; ks < nks; ++ks) {
    int buf = ks & 1;
    if (ks + 1 < nks) { stage(ks + 1, buf ^ 1); async_wait2(); }
    else              { async_wait0(); }
    __syncthreads();
    #pragma unroll
    for (int i = 0; i < 2; ++i) {
      int t = wave * 2 + i;
      int mt = t >> 2, nt = t & 3;
      ABf af, bf;
      const uint32_t* ap = &sA[buf][(mt * 16 + l15) * 16];
      #pragma unroll
      for (int j = 0; j < 4; ++j) { af.u[j] = ap[kh2 + j]; af.u[4 + j] = ap[kh2 + 4 + j]; }
      #pragma unroll
      for (int v = 0; v < 8; ++v) bf.u[v] = sB[buf][(kbh2 + v) * 64 + nt * 16 + l15];
      acc[i] = wmma_bf(af, bf, acc[i]);
    }
    __syncthreads();
  }

  #pragma unroll
  for (int i = 0; i < 2; ++i) {
    int t = wave * 2 + i;
    int mt = t >> 2, nt = t & 3;
    int n = bn * 64 + nt * 16 + l15;
    float bv = bias[n];
    int mb = bm * 64 + mt * 16 + hi * 8;
    #pragma unroll
    for (int r = 0; r < 8; ++r) {
      float val = acc[i][r] + bv;
      size_t idx = (size_t)(mb + r) * N + n;
      if (mode == 0) obf[idx] = f2bf(gelu_f(val));
      else           of32[idx] += val;
    }
  }
}

// =====================================================================
// Depthwise 3x3 conv (HID=768 channels) + exact GELU, one batch image
// =====================================================================
__global__ __launch_bounds__(256) void dwconv_gelu(const uint16_t* __restrict__ h1,
                                                   const float* __restrict__ w,
                                                   const float* __restrict__ bias,
                                                   uint16_t* __restrict__ outb) {
  int i = blockIdx.x * 256 + threadIdx.x;
  if (i >= 16384 * 768) return;
  int c = i % 768, p = i / 768;
  int y = p >> 7, x = p & 127;
  float s = bias[c];
  #pragma unroll
  for (int dy = -1; dy <= 1; ++dy)
    #pragma unroll
    for (int dx = -1; dx <= 1; ++dx) {
      int yy = y + dy, xx = x + dx;
      if (yy >= 0 && yy < 128 && xx >= 0 && xx < 128)
        s += w[c * 9 + (dy + 1) * 3 + (dx + 1)] * bf2f(h1[(size_t)((yy << 7) + xx) * 768 + c]);
    }
  outb[(size_t)i] = f2bf(gelu_f(s));
}

// =====================================================================
// Orchestration
// =====================================================================
extern "C" void kernel_launch(void* const* d_in, const int* in_sizes, int n_in,
                              void* d_out, int out_size, void* d_ws, size_t ws_size,
                              hipStream_t stream) {
  (void)in_sizes; (void)n_in; (void)out_size; (void)ws_size;
  const float* x      = (const float*)d_in[0];
  const float* src    = (const float*)d_in[1];
  const float* ln1_g  = (const float*)d_in[2];
  const float* ln1_b  = (const float*)d_in[3];
  const float* s1_w1  = (const float*)d_in[4];
  const float* s1_b1  = (const float*)d_in[5];
  const float* s1_wg  = (const float*)d_in[6];
  const float* s1_bg  = (const float*)d_in[7];
  const float* s1_wb  = (const float*)d_in[8];
  const float* s1_bb  = (const float*)d_in[9];
  const float* qkv_w  = (const float*)d_in[10];
  const float* qkv_b  = (const float*)d_in[11];
  const float* rpb    = (const float*)d_in[12];
  const float* proj_w = (const float*)d_in[13];
  const float* proj_b = (const float*)d_in[14];
  const float* ln2_g  = (const float*)d_in[15];
  const float* ln2_b  = (const float*)d_in[16];
  const float* s2_w1  = (const float*)d_in[17];
  const float* s2_b1  = (const float*)d_in[18];
  const float* s2_wg  = (const float*)d_in[19];
  const float* s2_bg  = (const float*)d_in[20];
  const float* s2_wb  = (const float*)d_in[21];
  const float* s2_bb  = (const float*)d_in[22];
  const float* l1_w   = (const float*)d_in[23];
  const float* l1_b   = (const float*)d_in[24];
  const float* dw_w   = (const float*)d_in[25];
  const float* dw_b   = (const float*)d_in[26];
  const float* l2_w   = (const float*)d_in[27];
  const float* l2_b   = (const float*)d_in[28];
  float* out = (float*)d_out;
  char* ws = (char*)d_ws;

  size_t off = 0;
  auto take = [&](size_t bytes) -> char* {
    char* p = ws + off; off += (bytes + 255) & ~(size_t)255; return p;
  };
  uint16_t* pk_qkv  = (uint16_t*)take((size_t)192 * 576 * 2);
  uint16_t* pk_proj = (uint16_t*)take((size_t)192 * 192 * 2);
  uint16_t* pk_l1   = (uint16_t*)take((size_t)192 * 768 * 2);
  uint16_t* pk_l2   = (uint16_t*)take((size_t)768 * 192 * 2);
  uint16_t* pk_wg1  = (uint16_t*)take((size_t)1152 * 192 * 2);
  uint16_t* pk_wb1  = (uint16_t*)take((size_t)1152 * 192 * 2);
  uint16_t* pk_wg2  = (uint16_t*)take((size_t)1152 * 192 * 2);
  uint16_t* pk_wb2  = (uint16_t*)take((size_t)1152 * 192 * 2);
  uint16_t* bufA    = (uint16_t*)take((size_t)B_ * L_ * 128 * 2);   // 33.5 MB
  uint16_t* bufG    = (uint16_t*)take((size_t)B_ * L_ * C_ * 2);    // 50.3 MB
  uint16_t* bufBt   = (uint16_t*)take((size_t)B_ * L_ * C_ * 2);    // 50.3 MB
  uint16_t* bufXW   = (uint16_t*)take((size_t)B_ * L_ * C_ * 2);    // 50.3 MB
  uint16_t* bufH1   = (uint16_t*)take((size_t)L_ * HID_ * 2);       // 25.2 MB (per-batch)
  uint16_t* bufDW   = (uint16_t*)take((size_t)L_ * HID_ * 2);       // 25.2 MB (per-batch)

  auto g1 = [](int n) { return (n + 255) / 256; };

  // weight packing (bf16, pair-interleaved)
  pack_rowmajor<<<g1(192 * 576), 256, 0, stream>>>(qkv_w,  pk_qkv,  192, 576);
  pack_rowmajor<<<g1(192 * 192), 256, 0, stream>>>(proj_w, pk_proj, 192, 192);
  pack_rowmajor<<<g1(192 * 768), 256, 0, stream>>>(l1_w,   pk_l1,   192, 768);
  pack_rowmajor<<<g1(768 * 192), 256, 0, stream>>>(l2_w,   pk_l2,   768, 192);
  pack_conv<<<g1(1152 * 192), 256, 0, stream>>>(s1_wg, pk_wg1);
  pack_conv<<<g1(1152 * 192), 256, 0, stream>>>(s1_wb, pk_wb1);
  pack_conv<<<g1(1152 * 192), 256, 0, stream>>>(s2_wg, pk_wg2);
  pack_conv<<<g1(1152 * 192), 256, 0, stream>>>(s2_wb, pk_wb2);

  // SPADE 1 (cubic resize to identical shape == identity, seg = src)
  conv1_lrelu<<<1024, 256, 0, stream>>>(src, s1_w1, s1_b1, bufA);
  conv_gb<<<2048, 128, 0, stream>>>(bufA, pk_wg1, s1_bg, bufG);
  conv_gb<<<2048, 128, 0, stream>>>(bufA, pk_wb1, s1_bb, bufBt);
  spade_mod<<<16384, 256, 0, stream>>>(x, bufG, bufBt, ln1_g, ln1_b, bufXW, 1);

  // shifted-window attention + residual -> d_out = x1
  attn_win<<<2048, 256, 0, stream>>>(bufXW, pk_qkv, qkv_b, rpb, pk_proj, proj_b, x, out);

  // SPADE 2 on x1
  conv1_lrelu<<<1024, 256, 0, stream>>>(src, s2_w1, s2_b1, bufA);
  conv_gb<<<2048, 128, 0, stream>>>(bufA, pk_wg2, s2_bg, bufG);
  conv_gb<<<2048, 128, 0, stream>>>(bufA, pk_wb2, s2_bb, bufBt);
  spade_mod<<<16384, 256, 0, stream>>>(out, bufG, bufBt, ln2_g, ln2_b, bufXW, 0);

  // LeFF, one batch image at a time (spatially independent) to bound scratch
  for (int b = 0; b < B_; ++b) {
    const uint16_t* h2b = bufXW + (size_t)b * L_ * C_;
    gemm_bf16<<<256 * 12, 256, 0, stream>>>(h2b, pk_l1, l1_b, bufH1, nullptr,
                                            L_, HID_, C_, 0);
    dwconv_gelu<<<49152, 256, 0, stream>>>(bufH1, dw_w, dw_b, bufDW);
    gemm_bf16<<<256 * 3, 256, 0, stream>>>(bufDW, pk_l2, l2_b, nullptr,
                                           out + (size_t)b * L_ * C_,
                                           L_, C_, HID_, 1);
  }
}
